// DepthLSSTransform_87273735454764
// MI455X (gfx1250) — compile-verified, matrix operand order checked by reading, not run on hardware
//
#include <hip/hip_runtime.h>
#include <hip/hip_fp16.h>

typedef __attribute__((ext_vector_type(16))) _Float16 v16h;
typedef __attribute__((ext_vector_type(8)))  _Float16 v8h;
typedef __attribute__((ext_vector_type(8)))  float    v8f;
typedef __attribute__((ext_vector_type(4)))  unsigned int v4u;
typedef __attribute__((ext_vector_type(8)))  int v8i;
typedef __attribute__((ext_vector_type(4)))  int v4i;

#define Bn   2
#define Nv   6
#define Gg   4
#define Pp   8
#define CINc 256
#define COUTc 128
#define CIMGc 32
#define GXg  64
#define GYg  64
#define GZg  4
#define Qq   8192

#if defined(__gfx1250__) && __has_builtin(__builtin_amdgcn_tensor_load_to_lds) && \
    __has_builtin(__builtin_amdgcn_s_wait_tensorcnt)
#define USE_TDM 1
#else
#define USE_TDM 0
#endif

// ---------------- WMMA fragment helpers (16x16x32 f16 -> f32) ----------------
// A (16x32 f16): lane = {row = lane&15, half = lane>>4}. Per lane the 16 halves
// are K = half*8 + [0..7]  and  K = 16 + half*8 + [0..7]  (two 16B chunks).
// B (32x16 f16) stored pre-transposed as Bt[N][K] so the same loader applies
// with lane&15 selecting the N column-row.
union FragU { v16h v; v8h h[2]; };

__device__ __forceinline__ v16h frag_load(const _Float16* __restrict__ rowbase, bool pred) {
  v16h out = {};
  if (pred) {
    const int hh = (threadIdx.x >> 4) & 1;
    FragU f;
    f.h[0] = *(const v8h*)(rowbase + hh * 8);
    f.h[1] = *(const v8h*)(rowbase + 16 + hh * 8);
    out = f.v;
  }
  return out;
}

__device__ __forceinline__ v16h frag_load_lds(const _Float16* rowbase) {
  const int hh = (threadIdx.x >> 4) & 1;
  FragU f;
  f.h[0] = *(const v8h*)(rowbase + hh * 8);
  f.h[1] = *(const v8h*)(rowbase + 16 + hh * 8);
  return f.v;
}

__device__ __forceinline__ v8f wmma16(v16h a, v16h b, v8f c) {
  return __builtin_amdgcn_wmma_f32_16x16x32_f16(false, a, false, b, (short)0, c, false, false);
}

// ---------------- conversion / utility kernels ----------------
__global__ void k_zero(float* __restrict__ p, int n) {
  int t = blockIdx.x * blockDim.x + threadIdx.x;
  if (t < n) p[t] = 0.f;
}

// src [K][N] f32 row-major -> dst [N][K] f16
__global__ void k_transpose_cvt(const float* __restrict__ src, _Float16* __restrict__ dst,
                                int K, int N) {
  int t = blockIdx.x * blockDim.x + threadIdx.x;
  if (t >= K * N) return;
  int k = t / N, n = t - k * N;
  dst[(size_t)n * K + k] = (_Float16)src[t];
}

// conv weight [Co][Ci][3][3] f32 -> [Co][tap*Ci + ci] f16 (tap-major K)
__global__ void k_conv_w_cvt(const float* __restrict__ src, _Float16* __restrict__ dst,
                             int Cin, int Cout) {
  int t = blockIdx.x * blockDim.x + threadIdx.x;
  int tot = Cout * Cin * 9;
  if (t >= tot) return;
  int co = t / (Cin * 9);
  int r  = t - co * (Cin * 9);
  int ci = r / 9;
  int kt = r - ci * 9;
  dst[(size_t)co * (Cin * 9) + kt * Cin + ci] = (_Float16)src[t];
}

// gather BEV point features per query -> f16 [16384][256]
__global__ void k_gather(const float* __restrict__ pts_bev,
                         const int* __restrict__ vx, const int* __restrict__ vy,
                         _Float16* __restrict__ ptsf) {
  int row = blockIdx.x;             // 0..16383
  int c   = threadIdx.x;            // 0..255
  int b   = row >> 13;
  int X = vx[row], Y = vy[row];
  float v = pts_bev[(((size_t)(b * CINc + c)) * GYg + Y) * GXg + X];
  ptsf[(size_t)row * CINc + c] = (_Float16)v;
}

// ---------------- GEMM 1: pts_f @ [w_offset | w_scale]  (K=256, N=224) -------
__global__ void k_gemm_os(const _Float16* __restrict__ A, const _Float16* __restrict__ Bt,
                          float* __restrict__ Craw) {
  const int mt   = blockIdx.x;          // 1024 M-tiles of 16
  const int wave = threadIdx.x >> 5;
  const int lane = threadIdx.x & 31;
  const int r15  = lane & 15, hh = lane >> 4;
  const _Float16* arow = A + ((size_t)mt * 16 + r15) * CINc;
  v16h af[8];
#pragma unroll
  for (int k = 0; k < 8; ++k) af[k] = frag_load(arow + k * 32, true);
  for (int nt = wave; nt < 14; nt += 8) {
    const _Float16* brow = Bt + ((size_t)nt * 16 + r15) * CINc;
    v8f acc = {};
#pragma unroll
    for (int k = 0; k < 8; ++k) {
      v16h bf = frag_load(brow + k * 32, true);
      acc = wmma16(af[k], bf, acc);
    }
    float* out = Craw + (size_t)mt * 16 * 224 + nt * 16;
#pragma unroll
    for (int r = 0; r < 8; ++r) out[(size_t)(r + 8 * hh) * 224 + r15] = acc[r];
  }
}

// ---------------- sampling: offsets, softmax scales, projection, bilinear ----
__global__ void k_sample(const float* __restrict__ f0, const float* __restrict__ f1,
                         const float* __restrict__ f2, const float* __restrict__ f3,
                         const float* __restrict__ rawos,
                         const float* __restrict__ boff, const float* __restrict__ bsc,
                         const float* __restrict__ l2i,
                         const int* __restrict__ vx, const int* __restrict__ vy,
                         const int* __restrict__ vz,
                         _Float16* __restrict__ outh) {
  int t = blockIdx.x * blockDim.x + threadIdx.x;
  if (t >= Bn * Qq * Gg * Pp) return;
  const int gp  = t & 31;
  const int row = t >> 5;
  const int g = gp >> 3, p = gp & 7;
  const int b = row >> 13;
  const float* ro = rawos + (size_t)row * 224;
  float offv[3];
#pragma unroll
  for (int d = 0; d < 3; ++d) offv[d] = ro[gp * 3 + d] + boff[gp * 3 + d];
  float sc[4];
  float mx = -1e30f;
#pragma unroll
  for (int l = 0; l < 4; ++l) { sc[l] = ro[96 + gp * 4 + l] + bsc[gp * 4 + l]; mx = fmaxf(mx, sc[l]); }
  float se = 0.f;
#pragma unroll
  for (int l = 0; l < 4; ++l) { sc[l] = __expf(sc[l] - mx); se += sc[l]; }
  float inv = 1.f / se;
#pragma unroll
  for (int l = 0; l < 4; ++l) sc[l] *= inv;

  const float sx = ((float)vx[row] + 0.5f) * 0.6f - 19.2f + offv[0] * 0.6f;
  const float sy = ((float)vy[row] + 0.5f) * 0.6f - 19.2f + offv[1] * 0.6f;
  const float sz = ((float)vz[row] + 0.5f) * 0.4f - 1.0f  + offv[2] * 0.4f;

  float u = 0.f, v = 0.f;
  int view = 0, found = 0;
#pragma unroll
  for (int n = 0; n < 6; ++n) {
    const float* M = l2i + (size_t)(b * 6 + n) * 16;
    float c0 = M[0] * sx + M[1] * sy + M[2]  * sz + M[3];
    float c1 = M[4] * sx + M[5] * sy + M[6]  * sz + M[7];
    float c2 = M[8] * sx + M[9] * sy + M[10] * sz + M[11];
    float hm = fmaxf(c2, 1e-5f);
    float uu = c0 / (hm * 704.f);
    float vv = c1 / (hm * 256.f);
    int val = (c2 > 1e-5f) & (uu > 0.f) & (uu < 1.f) & (vv > 0.f) & (vv < 1.f);
    if (n == 0) { u = uu; v = vv; }
    if (val && !found) { found = 1; view = n; u = uu; v = vv; }
  }

  float acc[CIMGc];
#pragma unroll
  for (int c = 0; c < CIMGc; ++c) acc[c] = 0.f;
  const float* fl[4] = { f0, f1, f2, f3 };
  const int HH[4] = { 32, 16, 8, 4 };
  const int WW[4] = { 88, 44, 22, 11 };
#pragma unroll
  for (int l = 0; l < 4; ++l) {
    const int H = HH[l], W = WW[l];
    const float* fp = fl[l] + ((size_t)(b * Nv + view) * 128 + g * 32) * H * W;
    float px = u * (float)W - 0.5f, py = v * (float)H - 0.5f;
    float x0f = floorf(px), y0f = floorf(py);
    float wx = px - x0f, wy = py - y0f;
    int x0 = (int)x0f, y0 = (int)y0f;
    float cw[4] = { (1.f - wx) * (1.f - wy), wx * (1.f - wy), (1.f - wx) * wy, wx * wy };
    int cx[4] = { x0, x0 + 1, x0, x0 + 1 };
    int cy[4] = { y0, y0, y0 + 1, y0 + 1 };
#pragma unroll
    for (int k = 0; k < 4; ++k) {
      int xi = cx[k], yi = cy[k];
      if (xi < 0 || xi >= W || yi < 0 || yi >= H) continue;
      float wgt = sc[l] * cw[k];
      const float* q = fp + (size_t)yi * W + xi;
#pragma unroll
      for (int c = 0; c < CIMGc; ++c) acc[c] += wgt * q[(size_t)c * H * W];
    }
  }
  // final layout [B,Q,P,G,CIMG]
  _Float16* o = outh + (((size_t)row * Pp + p) * Gg + g) * CIMGc;
#pragma unroll
  for (int c = 0; c < CIMGc; ++c) o[c] = (_Float16)acc[c];
}

// ---------------- GEMM 2: final[16384,1024] @ w_aggT -> scatter into BEV -----
// A tile (16 rows x 1024 halves) is one contiguous 32KB block: stage it into
// dynamic LDS with one Tensor Data Mover descriptor per workgroup, then all
// 8 waves read A fragments from LDS (B stays in global, served by L2).
__global__ void k_gemm_agg(const _Float16* __restrict__ A, const _Float16* __restrict__ Bt,
                           const float* __restrict__ bagg, const float* __restrict__ qmask,
                           const int* __restrict__ vx, const int* __restrict__ vy,
                           const int* __restrict__ vz, float* __restrict__ ffeat) {
  const int mt   = blockIdx.x;          // 1024 M-tiles
  const int nt   = threadIdx.x >> 5;    // 8 N-tiles, one per wave
  const int lane = threadIdx.x & 31;
  const int r15  = lane & 15, hh = lane >> 4;

  extern __shared__ _Float16 ldsA[];    // 32KB dynamic LDS

#if USE_TDM
  if (threadIdx.x < 32) {
    // LDS byte offset of the dynamic-shared base (aperture rule: addr[31:0])
    unsigned ldsoff = (unsigned)(unsigned long long)(void*)ldsA;
    // D# group 0: count=1 | lds_addr | global_addr | type=2 (image)
    unsigned long long ga = (unsigned long long)(A + (size_t)mt * 16 * 1024);
    v4u g0 = { 1u,                                   // count=1, user mode
               ldsoff,                               // lds_addr
               (unsigned)(ga & 0xffffffffull),
               (unsigned)(((ga >> 32) & 0x01ffffffull) | (2u << 30)) };
    // D# group 1: data_size=8B; 1D tile of 4096 elements (= 32KB)
    v8i g1 = { (int)(3u << 16),                      // wg_mask=0, data_size=3 (8B)
               (int)(4096u << 16),                   // tensor_dim0[15:0] << 16
               (int)((4096u >> 16) | (1u << 16)),    // tensor_dim0 hi | tensor_dim1=1
               (int)(4096u << 16),                   // tile_dim0 = 4096
               1,                                    // tile_dim1 = 1
               4096,                                 // tensor_dim0_stride lo32
               0, 0 };
    v4i z4 = {};
    v8i z8 = {};
    __builtin_amdgcn_tensor_load_to_lds(g0, g1, z4, z4, z8, 0);
    __builtin_amdgcn_s_wait_tensorcnt(0);
  }
  __syncthreads();
  asm volatile("" ::: "memory");        // DMA wrote LDS behind the compiler's back
  const _Float16* arow = ldsA + (size_t)r15 * 1024;
#else
  // fallback: cooperative global->LDS staging
  {
    const _Float16* src = A + (size_t)mt * 16 * 1024;
    for (int i = threadIdx.x; i < 16 * 1024 / 8; i += blockDim.x)
      *(v8h*)(ldsA + i * 8) = *(const v8h*)(src + i * 8);
  }
  __syncthreads();
  const _Float16* arow = ldsA + (size_t)r15 * 1024;
#endif

  const _Float16* brow = Bt + ((size_t)nt * 16 + r15) * 1024;
  v8f acc = {};
#pragma unroll 4
  for (int k0 = 0; k0 < 1024; k0 += 32) {
    __builtin_prefetch(brow + k0 + 256, 0, 1);
    v16h a = frag_load_lds(arow + k0);
    v16h b = frag_load(brow + k0, true);
    acc = wmma16(a, b, acc);
  }
  const int co = nt * 16 + r15;
  const float bias = bagg[co];
#pragma unroll
  for (int r = 0; r < 8; ++r) {
    int row = mt * 16 + r + 8 * hh;
    int b_  = row >> 13;
    float val = (acc[r] + bias) * qmask[row];
    int X = vx[row], Y = vy[row], Z = vz[row];
    atomicAdd(&ffeat[((((size_t)b_ * GZg + Z) * GYg + Y) * GXg + X) * COUTc + co], val);
  }
}

// ffeat [B][GZ][GY][GX][COUT] f32 -> act0 NHWC f16 with channel = c*GZ + z
__global__ void k_ffeat_to_act0(const float* __restrict__ ffeat, _Float16* __restrict__ act0) {
  int t = blockIdx.x * blockDim.x + threadIdx.x;
  if (t >= Bn * GZg * GYg * GXg * COUTc) return;
  int c  = t & 127;
  int r  = t >> 7;
  int x  = r & 63; r >>= 6;
  int y  = r & 63; r >>= 6;
  int z  = r & 3;  r >>= 2;
  int b  = r;
  act0[(((size_t)(b * 64 + y) * 64 + x) * 512) + c * 4 + z] = (_Float16)ffeat[t];
}

// ---------------- implicit-GEMM 3x3 conv + BN + ReLU (NHWC f16 in) ----------
template <int Cin, int Cout>
__global__ void k_conv(const _Float16* __restrict__ act, const _Float16* __restrict__ wT,
                       const float* __restrict__ bng, const float* __restrict__ bnb,
                       _Float16* __restrict__ out16, float* __restrict__ out32) {
  const int mt   = blockIdx.x;          // 512 M-tiles of 16 pixels
  const int wave = threadIdx.x >> 5;
  const int lane = threadIdx.x & 31;
  const int r15  = lane & 15, hh = lane >> 4;
  const int m0   = mt * 16;
  const int b    = m0 >> 12;
  const int rem  = m0 & 4095;
  const int y    = rem >> 6;
  const int x0   = rem & 63;            // multiple of 16
  const int px   = x0 + r15;            // this lane's A-row pixel x
  constexpr int K9 = Cin * 9;
  constexpr int ntiles = Cout >> 4;
  const float bnscale = 0.9999950000374997f;   // 1/sqrt(1+1e-5)

  for (int nt = wave; nt < ntiles; nt += 8) {
    const _Float16* brow = wT + ((size_t)nt * 16 + r15) * K9;
    v8f acc = {};
#pragma unroll
    for (int ty = -1; ty <= 1; ++ty) {
      int yy = y + ty;
      bool yok = (yy >= 0) && (yy < 64);
#pragma unroll
      for (int tx = -1; tx <= 1; ++tx) {
        int xx = px + tx;
        bool ok = yok && (xx >= 0) && (xx < 64);
        const _Float16* apix = act + ((size_t)(b * 64 + yy) * 64 + xx) * Cin;
        const int tap = (ty + 1) * 3 + (tx + 1);
        const _Float16* btap = brow + (size_t)tap * Cin;
#pragma unroll 4
        for (int c0 = 0; c0 < Cin; c0 += 32) {
          v16h a = frag_load(apix + c0, ok);
          v16h bb = frag_load(btap + c0, true);
          acc = wmma16(a, bb, acc);
        }
      }
    }
    const int co = nt * 16 + r15;
    const float s  = bng[co] * bnscale;
    const float bi = bnb[co];
#pragma unroll
    for (int r = 0; r < 8; ++r) {
      int ml = r + 8 * hh;
      int xx = x0 + ml;
      float v = acc[r] * s + bi;
      v = v > 0.f ? v : 0.f;
      if (out32) out32[(((size_t)b * Cout + co) * 64 + y) * 64 + xx] = v;
      else       out16[(((size_t)(b * 64 + y) * 64 + xx)) * Cout + co] = (_Float16)v;
    }
  }
}

// ---------------------------------------------------------------------------
extern "C" void kernel_launch(void* const* d_in, const int* in_sizes, int n_in,
                              void* d_out, int out_size, void* d_ws, size_t ws_size,
                              hipStream_t stream) {
  (void)in_sizes; (void)n_in; (void)out_size; (void)ws_size;
  const float* feat0    = (const float*)d_in[0];
  const float* feat1    = (const float*)d_in[1];
  const float* feat2    = (const float*)d_in[2];
  const float* feat3    = (const float*)d_in[3];
  const float* pts_bev  = (const float*)d_in[4];
  const float* l2i      = (const float*)d_in[5];
  const float* qmask    = (const float*)d_in[6];
  const int*   vox_x    = (const int*)d_in[7];
  const int*   vox_y    = (const int*)d_in[8];
  const int*   vox_z    = (const int*)d_in[9];
  const float* w_offset = (const float*)d_in[10];
  const float* b_offset = (const float*)d_in[11];
  const float* w_scale  = (const float*)d_in[12];
  const float* b_scale  = (const float*)d_in[13];
  const float* w_agg    = (const float*)d_in[14];
  const float* b_agg    = (const float*)d_in[15];
  const float* conv1_w  = (const float*)d_in[16];
  const float* conv2_w  = (const float*)d_in[17];
  const float* conv3_w  = (const float*)d_in[18];
  const float* conv4_w  = (const float*)d_in[19];
  const float* bn1_g = (const float*)d_in[20]; const float* bn1_b = (const float*)d_in[21];
  const float* bn2_g = (const float*)d_in[22]; const float* bn2_b = (const float*)d_in[23];
  const float* bn3_g = (const float*)d_in[24]; const float* bn3_b = (const float*)d_in[25];
  const float* bn4_g = (const float*)d_in[26]; const float* bn4_b = (const float*)d_in[27];
  float* out = (float*)d_out;

  char* ws = (char*)d_ws;
  size_t o = 0;
  auto alloc = [&](size_t bytes) {
    char* p = ws + o;
    o = (o + bytes + 255) & ~(size_t)255;
    return p;
  };
  _Float16* wsOS   = (_Float16*)alloc((size_t)224 * 256 * 2);
  _Float16* waggT  = (_Float16*)alloc((size_t)128 * 1024 * 2);
  _Float16* wc1T   = (_Float16*)alloc((size_t)256 * 512 * 9 * 2);
  _Float16* wc2T   = (_Float16*)alloc((size_t)512 * 256 * 9 * 2);
  _Float16* wc3T   = (_Float16*)alloc((size_t)512 * 512 * 9 * 2);
  _Float16* wc4T   = (_Float16*)alloc((size_t)128 * 512 * 9 * 2);
  _Float16* ptsf   = (_Float16*)alloc((size_t)16384 * 256 * 2);
  float*    rawos  = (float*)   alloc((size_t)16384 * 224 * 4);
  _Float16* finalh = (_Float16*)alloc((size_t)16384 * 1024 * 2);
  float*    ffeat  = (float*)   alloc((size_t)Bn * GZg * GYg * GXg * COUTc * 4);
  _Float16* act0   = (_Float16*)alloc((size_t)Bn * 64 * 64 * 512 * 2);
  _Float16* act1   = (_Float16*)alloc((size_t)Bn * 64 * 64 * 256 * 2);
  _Float16* act2   = (_Float16*)alloc((size_t)Bn * 64 * 64 * 512 * 2);
  _Float16* act3   = (_Float16*)alloc((size_t)Bn * 64 * 64 * 512 * 2);

  dim3 blk(256);
  auto gs = [](size_t n) { return dim3((unsigned)((n + 255) / 256)); };

  // weight conversion / transposition
  k_transpose_cvt<<<gs(256 * 96),  blk, 0, stream>>>(w_offset, wsOS, 256, 96);
  k_transpose_cvt<<<gs(256 * 128), blk, 0, stream>>>(w_scale,  wsOS + (size_t)96 * 256, 256, 128);
  k_transpose_cvt<<<gs(1024 * 128), blk, 0, stream>>>(w_agg, waggT, 1024, 128);
  k_conv_w_cvt<<<gs((size_t)256 * 512 * 9), blk, 0, stream>>>(conv1_w, wc1T, 512, 256);
  k_conv_w_cvt<<<gs((size_t)512 * 256 * 9), blk, 0, stream>>>(conv2_w, wc2T, 256, 512);
  k_conv_w_cvt<<<gs((size_t)512 * 512 * 9), blk, 0, stream>>>(conv3_w, wc3T, 512, 512);
  k_conv_w_cvt<<<gs((size_t)128 * 512 * 9), blk, 0, stream>>>(conv4_w, wc4T, 512, 128);

  // gather + offset/scale GEMM
  k_gather<<<dim3(16384), blk, 0, stream>>>(pts_bev, vox_x, vox_y, ptsf);
  k_gemm_os<<<dim3(1024), blk, 0, stream>>>(ptsf, wsOS, rawos);

  // projection + multi-level bilinear sampling
  k_sample<<<gs((size_t)Bn * Qq * Gg * Pp), blk, 0, stream>>>(
      feat0, feat1, feat2, feat3, rawos, b_offset, b_scale, l2i,
      vox_x, vox_y, vox_z, finalh);

  // aggregation GEMM (TDM-staged A tile in 32KB dynamic LDS) + scatter
  k_zero<<<gs((size_t)Bn * GZg * GYg * GXg * COUTc), blk, 0, stream>>>(
      ffeat, Bn * GZg * GYg * GXg * COUTc);
  k_gemm_agg<<<dim3(1024), blk, 32768, stream>>>(finalh, waggT, b_agg, qmask,
                                                 vox_x, vox_y, vox_z, ffeat);
  k_ffeat_to_act0<<<gs((size_t)Bn * GZg * GYg * GXg * COUTc), blk, 0, stream>>>(ffeat, act0);

  // conv stack (implicit GEMM WMMA, compile-time channel counts)
  k_conv<512, 256><<<dim3(512), blk, 0, stream>>>(act0, wc1T, bn1_g, bn1_b, act1, nullptr);
  k_conv<256, 512><<<dim3(512), blk, 0, stream>>>(act1, wc2T, bn2_g, bn2_b, act2, nullptr);
  k_conv<512, 512><<<dim3(512), blk, 0, stream>>>(act2, wc3T, bn3_g, bn3_b, act3, nullptr);
  k_conv<512, 128><<<dim3(512), blk, 0, stream>>>(act3, wc4T, bn4_g, bn4_b, nullptr, out);
}
